// rnnnet_47064251629795
// MI455X (gfx1250) — matrix operand order repaired and checked by reading, not compiled, not run on hardware
//
#include <hip/hip_runtime.h>
#include <hip/hip_bf16.h>
#include <math.h>

typedef __bf16 bf16;
typedef bf16  v16bf __attribute__((ext_vector_type(16)));
typedef float v8f   __attribute__((ext_vector_type(8)));

union V16 { v16bf v; uint4 u[2]; };

#define T_STEPS 2048
#define BATCH   1024
#define IDIM    32
#define HDIM    128

static __device__ __forceinline__ unsigned int pack2bf(float2 f) {
    bf16 a = (bf16)f.x, b = (bf16)f.y;
    unsigned short ua, ub;
    __builtin_memcpy(&ua, &a, 2);
    __builtin_memcpy(&ub, &b, 2);
    return ((unsigned int)ub << 16) | (unsigned int)ua;
}

// Fast tanh: prefer the gfx1250 V_TANH_F32 transcendental (single TRANS op,
// co-executes with WMMA); otherwise a branch-free exp2/rcp formulation
// (~5 VALU, no lane divergence) instead of libm's branchy expansion.
static __device__ __forceinline__ float fast_tanh(float x) {
#if __has_builtin(__builtin_amdgcn_tanhf)
    return __builtin_amdgcn_tanhf(x);
#elif __has_builtin(__builtin_amdgcn_tanh_f32)
    return __builtin_amdgcn_tanh_f32(x);
#else
    // tanh(x) = 1 - 2/(exp(2x)+1); clamp so exp2 stays finite.
    float xc = fminf(fmaxf(x, -9.0f), 9.0f);
    float e  = __builtin_amdgcn_exp2f(xc * 2.88539008177792681472f); // 2*log2(e)
    return 1.0f - 2.0f * __builtin_amdgcn_rcpf(e + 1.0f);
#endif
}

// ---------------- prep: convert weights to bf16, fuse biases ----------------
__global__ void rnn_prep(const float* __restrict__ W_ih, const float* __restrict__ W_hh,
                         const float* __restrict__ W_out, const float* __restrict__ b_ih,
                         const float* __restrict__ b_hh, void* __restrict__ ws) {
    bf16* whh  = (bf16*)ws;                 // [128][128] row-major (= B-operand col-major in K)
    bf16* wih  = whh + HDIM * HDIM;         // [128][32]  row-major
    bf16* wout = wih + HDIM * IDIM;         // [128]
    float* bias = (float*)(wout + HDIM);    // [128] = b_ih + b_hh
    int i = blockIdx.x * blockDim.x + threadIdx.x;
    int stride = gridDim.x * blockDim.x;
    for (int j = i; j < HDIM * HDIM; j += stride) whh[j] = (bf16)W_hh[j];
    for (int j = i; j < HDIM * IDIM; j += stride) wih[j] = (bf16)W_ih[j];
    for (int j = i; j < HDIM; j += stride) {
        wout[j] = (bf16)W_out[j];
        bias[j] = b_ih[j] + b_hh[j];
    }
}

// ---------------- main persistent fused RNN kernel ----------------
// grid = B/16 workgroups; block = 256 threads (8 wave32's).
// Each WG owns 16 batch rows for all T steps. Each wave owns a 16-wide slice
// of the H=128 output columns. Weights live in VGPRs; h lives in LDS (bf16,
// double buffered); one s_barrier per timestep.
__global__ __launch_bounds__(256) void rnn_main(const float* __restrict__ x,
                                                const float* __restrict__ h0,
                                                const void*  __restrict__ ws,
                                                const float* __restrict__ b_out_p,
                                                float* __restrict__ out) {
    __shared__ bf16 hbuf[2][16][HDIM];          // h_t tiles, double buffered (8 KB)
    __shared__ unsigned int xbufu[2][256];      // x_t tile as packed bf16 pairs (2 KB)

    const int tid  = threadIdx.x;
    const int w    = tid >> 5;        // wave id 0..7
    const int lane = tid & 31;
    const int l16  = lane & 15;
    const int hi   = lane >> 4;
    const int n0   = w * 16;          // this wave's H-column base
    const int b0   = blockIdx.x * 16; // this WG's batch-row base

    const bf16*  whh  = (const bf16*)ws;
    const bf16*  wih  = whh + HDIM * HDIM;
    const bf16*  wout = wih + HDIM * IDIM;
    const float* bias = (const float*)(wout + HDIM);

    // ---- B operands, loaded once, register-resident for all 2048 steps ----
    // B[k][n] layout (bf16 32x16): lane l16 = column n0+l16; VGPRs hold 16
    // consecutive K values starting at 16*hi -> 32 contiguous bytes per lane.
    V16 Bhh[4];
#pragma unroll
    for (int kk = 0; kk < 4; ++kk) {
        const bf16* p = whh + (size_t)(n0 + l16) * HDIM + 32 * kk + 16 * hi;
        Bhh[kk].u[0] = *(const uint4*)(p);
        Bhh[kk].u[1] = *(const uint4*)(p + 8);
    }
    V16 Bih;
    {
        const bf16* p = wih + (size_t)(n0 + l16) * IDIM + 16 * hi;
        Bih.u[0] = *(const uint4*)(p);
        Bih.u[1] = *(const uint4*)(p + 8);
    }
    V16 By[4];  // W_out as a B operand: only column n=0 nonzero
#pragma unroll
    for (int kk = 0; kk < 4; ++kk) {
        By[kk].u[0] = uint4{0, 0, 0, 0};
        By[kk].u[1] = uint4{0, 0, 0, 0};
        if (l16 == 0) {
            const bf16* p = wout + 32 * kk + 16 * hi;
            By[kk].u[0] = *(const uint4*)(p);
            By[kk].u[1] = *(const uint4*)(p + 8);
        }
    }
    const float bv = bias[n0 + l16];
    const float bo = b_out_p[0];

    // ---- init h_0 into hbuf[0] (bf16) ----
    {
        bf16* hflat = &hbuf[0][0][0];
        const float* h0g = h0 + (size_t)b0 * HDIM;
        for (int j = tid; j < 16 * HDIM; j += 256) hflat[j] = (bf16)h0g[j];
    }

    // ---- prefetch x tile for t=0 (2 floats/thread; tile is contiguous) ----
    const float* xg = x + (size_t)b0 * IDIM + tid * 2;
    float2 xv = *(const float2*)(xg);

    float* yout = out;                               // y: [T,B]
    float* hid  = out + (size_t)T_STEPS * BATCH;     // hidden: [B,H]

#pragma unroll 2
    for (int t = 0; t < T_STEPS; ++t) {
        const int cur = t & 1, nxt = cur ^ 1;

        // stage x_t into LDS; prefetch x_{t+1} (latency overlaps this step)
        xbufu[cur][tid] = pack2bf(xv);
        if (t + 1 < T_STEPS)
            xv = *(const float2*)(xg + (size_t)(t + 1) * (BATCH * IDIM));

        __syncthreads();

        // A operand: x tile 16x32 bf16 (row-major rows of 32)
        const bf16* xb = (const bf16*)&xbufu[cur][0];
        V16 ax;
        ax.u[0] = *(const uint4*)(xb + l16 * IDIM + 16 * hi);
        ax.u[1] = *(const uint4*)(xb + l16 * IDIM + 16 * hi + 8);

        // A operand: h_t 16x128 bf16, four 16x32 K-chunks
        V16 ah[4];
#pragma unroll
        for (int kk = 0; kk < 4; ++kk) {
            const bf16* hr = &hbuf[cur][l16][32 * kk + 16 * hi];
            ah[kk].u[0] = *(const uint4*)(hr);
            ah[kk].u[1] = *(const uint4*)(hr + 8);
        }

        // C = x*W_ih^T + h*W_hh^T   (5 chained WMMAs)
        v8f c = {};
        c = __builtin_amdgcn_wmma_f32_16x16x32_bf16(false, ax.v, false, Bih.v,
                                                    (short)0, c, false, false);
#pragma unroll
        for (int kk = 0; kk < 4; ++kk)
            c = __builtin_amdgcn_wmma_f32_16x16x32_bf16(false, ah[kk].v, false, Bhh[kk].v,
                                                        (short)0, c, false, false);

        // y[t-1] = h_t . W_out + b_out, reusing the A operands just loaded.
        if (w == 0 && t > 0) {
            v8f cy = {};
#pragma unroll
            for (int kk = 0; kk < 4; ++kk)
                cy = __builtin_amdgcn_wmma_f32_16x16x32_bf16(false, ah[kk].v, false, By[kk].v,
                                                             (short)0, cy, false, false);
            if (l16 == 0) {  // column 0 of C holds the dot products
                float* yp = yout + (size_t)(t - 1) * BATCH + b0 + 8 * hi;
#pragma unroll
                for (int v = 0; v < 8; ++v) yp[v] = cy[v] + bo;
            }
        }

        // h_{t+1} = tanh(C + bias)
        float hn[8];
#pragma unroll
        for (int v = 0; v < 8; ++v) hn[v] = fast_tanh(c[v] + bv);

        // write h_{t+1} (bf16) into the other buffer
#pragma unroll
        for (int v = 0; v < 8; ++v)
            hbuf[nxt][v + 8 * hi][n0 + l16] = (bf16)hn[v];

        // final hidden state: full f32 straight from the accumulator
        if (t == T_STEPS - 1) {
#pragma unroll
            for (int v = 0; v < 8; ++v)
                hid[(size_t)(b0 + v + 8 * hi) * HDIM + n0 + l16] = hn[v];
        }
    }

    __syncthreads();

    // y[T-1] = h_T . W_out + b_out
    if (w == 0) {
        const int fin = T_STEPS & 1;  // buffer holding h_T
        v8f cy = {};
#pragma unroll
        for (int kk = 0; kk < 4; ++kk) {
            V16 ah;
            const bf16* hr = &hbuf[fin][l16][32 * kk + 16 * hi];
            ah.u[0] = *(const uint4*)(hr);
            ah.u[1] = *(const uint4*)(hr + 8);
            cy = __builtin_amdgcn_wmma_f32_16x16x32_bf16(false, ah.v, false, By[kk].v,
                                                         (short)0, cy, false, false);
        }
        if (l16 == 0) {
            float* yp = yout + (size_t)(T_STEPS - 1) * BATCH + b0 + 8 * hi;
#pragma unroll
            for (int v = 0; v < 8; ++v) yp[v] = cy[v] + bo;
        }
    }
}

extern "C" void kernel_launch(void* const* d_in, const int* in_sizes, int n_in,
                              void* d_out, int out_size, void* d_ws, size_t ws_size,
                              hipStream_t stream) {
    const float* x     = (const float*)d_in[0];
    const float* h0    = (const float*)d_in[1];
    const float* W_ih  = (const float*)d_in[2];
    const float* W_hh  = (const float*)d_in[3];
    const float* b_ih  = (const float*)d_in[4];
    const float* b_hh  = (const float*)d_in[5];
    const float* W_out = (const float*)d_in[6];
    const float* b_out = (const float*)d_in[7];
    (void)in_sizes; (void)n_in; (void)out_size; (void)ws_size;

    rnn_prep<<<64, 256, 0, stream>>>(W_ih, W_hh, W_out, b_ih, b_hh, d_ws);
    rnn_main<<<BATCH / 16, 256, 0, stream>>>(x, h0, d_ws, b_out, (float*)d_out);
}
